// AttentionBlock_88983132438550
// MI455X (gfx1250) — matrix operand order repaired and checked by reading, not compiled, not run on hardware
//
#include <hip/hip_runtime.h>
#include <hip/hip_bf16.h>

// ---------------------------------------------------------------------------
// AttentionBlock: GroupNorm -> QKV proj -> flash self-attention -> out proj
// B=4, H=W=64 (N=4096), C=512, 32 groups. bf16 WMMA (v_wmma_f32_16x16x32_bf16)
// with fp32 accumulation. Projection GEMMs stage their shared 64x512 weight
// tile in LDS via the Tensor Data Mover (tensor_load_to_lds + s_wait_tensorcnt)
// and read B-fragments with ds_load_b128. Flash attention keeps the 4096x4096
// score matrix entirely on-chip (online softmax).
// ---------------------------------------------------------------------------

typedef __attribute__((ext_vector_type(16))) __bf16 v16bf;
typedef __attribute__((ext_vector_type(8)))  __bf16 v8bf;
typedef __attribute__((ext_vector_type(8)))  float  v8f;
typedef __attribute__((ext_vector_type(4)))  unsigned int v4u;
typedef __attribute__((ext_vector_type(8)))  int v8i;
typedef __attribute__((ext_vector_type(4)))  int v4i;

#define CDIM  512
#define NDIM  4096
#define BATCH 4

__device__ __forceinline__ v16bf frag_cat(const __bf16* p0, const __bf16* p1) {
  v8bf lo = *(const v8bf*)p0;           // 16B -> global_load_b128 / ds_load_b128
  v8bf hi = *(const v8bf*)p1;
  return __builtin_shufflevector(lo, hi, 0,1,2,3,4,5,6,7,8,9,10,11,12,13,14,15);
}
// B-fragment: two 8-element bf16 runs along K, at p and p+16
__device__ __forceinline__ v16bf frag16(const __bf16* p) { return frag_cat(p, p + 16); }

__device__ __forceinline__ v8f wmma_bf16(v16bf a, v16bf b, v8f c) {
  return __builtin_amdgcn_wmma_f32_16x16x32_bf16(false, a, false, b, (short)0, c,
                                                 false, false);
}

// TDM: copy one contiguous 64KB region (8192 x 8B units) global -> LDS.
// D# built per CDNA5 ISA ch.8: group0 = {count=1, lds_addr, global_addr,
// type=2}; group1 = {data_size=3(8B), tensor_dim0=8192, tensor_dim1=1,
// tile_dim0=8192, stride0=8192}. Issued by one wave; EXEC is ignored by TDM.
// This toolchain exposes the 6-arg builtin (extra int32x8 group, then cpol).
__device__ __forceinline__ void tdm_load_64k(const void* gsrc, void* lds_dst) {
  unsigned ldsb = (unsigned)(size_t)lds_dst;          // low 32 bits = LDS offset
  unsigned long long ga = (unsigned long long)(size_t)gsrc;
  v4u g0 = { 1u,                                      // count=1 (valid D#)
             ldsb,                                    // lds_addr
             (unsigned)(ga & 0xFFFFFFFFu),            // global_addr[31:0]
             (unsigned)((ga >> 32) & 0x1FFFFFFu) | (2u << 30) }; // addr[56:32]|type=2
  v8i g1 = { (int)(3u << 16),                         // data_size = 8B
             (int)(8192u << 16),                      // tensor_dim0[15:0] << 16
             (int)(1u << 16),                         // tensor_dim0 hi=0 | tensor_dim1=1
             (int)(8192u << 16),                      // tensor_dim1 hi=0 | tile_dim0=8192
             0,                                       // tile_dim1=0, tile_dim2=0
             8192,                                    // tensor_dim0_stride[31:0]
             0, 0 };
  v4i gz4 = { 0, 0, 0, 0 };
  v8i gz8 = { 0, 0, 0, 0, 0, 0, 0, 0 };
  __builtin_amdgcn_tensor_load_to_lds(g0, g1, gz4, gz4, gz8, 0);
  __builtin_amdgcn_s_wait_tensorcnt(0);
}

// ---------------------------------------------------------------------------
// Kernel 1: convert fp32 weights [Cin][Cout] -> bf16 transposed [Cout][Cin]
// ---------------------------------------------------------------------------
__global__ __launch_bounds__(256) void prep_weights(
    const float* __restrict__ Wq, const float* __restrict__ Wk,
    const float* __restrict__ Wv, const float* __restrict__ Wo,
    __bf16* __restrict__ wt) {
  const int mat = blockIdx.y;
  const float* W = (mat == 0) ? Wq : (mat == 1) ? Wk : (mat == 2) ? Wv : Wo;
  const int idx  = blockIdx.x * 256 + threadIdx.x;   // 0..262143
  const int cout = idx >> 9;
  const int cin  = idx & 511;
  wt[(size_t)mat * CDIM * CDIM + idx] = (__bf16)W[(size_t)cin * CDIM + cout];
}

// ---------------------------------------------------------------------------
// Kernel 2: GroupNorm over (H,W,16ch) per (batch,group); output bf16 h[b][n][c]
// ---------------------------------------------------------------------------
__global__ __launch_bounds__(256) void group_norm_k(
    const float* __restrict__ x, const float* __restrict__ gamma,
    const float* __restrict__ beta, __bf16* __restrict__ h) {
  const int b = blockIdx.x >> 5;
  const int g = blockIdx.x & 31;
  const float* xp = x + (size_t)b * NDIM * CDIM + g * 16;
  float s = 0.f, s2 = 0.f;
  for (int i = threadIdx.x; i < NDIM * 16; i += 256) {
    const int n = i >> 4, cc = i & 15;
    const float v = xp[(size_t)n * CDIM + cc];
    s += v; s2 += v * v;
  }
  #pragma unroll
  for (int m = 16; m >= 1; m >>= 1) {
    s  += __shfl_xor(s,  m, 32);
    s2 += __shfl_xor(s2, m, 32);
  }
  __shared__ float rs[8], rs2[8];
  const int wid = threadIdx.x >> 5;
  if ((threadIdx.x & 31) == 0) { rs[wid] = s; rs2[wid] = s2; }
  __syncthreads();
  if (threadIdx.x == 0) {
    float a = 0.f, a2 = 0.f;
    for (int i = 0; i < 8; ++i) { a += rs[i]; a2 += rs2[i]; }
    rs[0] = a; rs2[0] = a2;
  }
  __syncthreads();
  const float inv_cnt = 1.0f / (float)(NDIM * 16);
  const float mean = rs[0] * inv_cnt;
  const float var  = rs2[0] * inv_cnt - mean * mean;
  const float inv  = rsqrtf(var + 1e-6f);
  for (int i = threadIdx.x; i < NDIM * 16; i += 256) {
    const int n = i >> 4, cc = i & 15, c = g * 16 + cc;
    const float v = xp[(size_t)n * CDIM + cc];
    h[((size_t)b * NDIM + n) * CDIM + c] =
        (__bf16)((v - mean) * inv * gamma[c] + beta[c]);
  }
}

// ---------------------------------------------------------------------------
// Kernel 3: Q/K/V projections. grid.z = {mat 0..2} x batch.
// Weight tile (64 couts x 512 cins, 64KB) staged in LDS via TDM, shared by
// all 8 waves. A fragments prefetched one K-step ahead.
// Q,K row-major bf16 [b][n][c]; V transposed bf16 [b][c][n].
// ---------------------------------------------------------------------------
__global__ __launch_bounds__(256) void qkv_gemm(
    const __bf16* __restrict__ h, const __bf16* __restrict__ wt,
    const float* __restrict__ bq, const float* __restrict__ bk,
    const float* __restrict__ bv, __bf16* __restrict__ Qg,
    __bf16* __restrict__ Kg, __bf16* __restrict__ Vtg) {
  __shared__ __bf16 sW[64 * CDIM];     // 64 KB weight tile
  const int mat = blockIdx.z % 3;
  const int b   = blockIdx.z / 3;
  const int lane = threadIdx.x & 31, w = threadIdx.x >> 5;
  const int r0 = blockIdx.x * 128 + w * 16;
  const int cbase = blockIdx.y * 64;
  const int ar = lane & 15;
  const int half = lane >> 4;
  const __bf16* A  = h + ((size_t)b * NDIM + r0) * CDIM;
  const __bf16* Wm = wt + (size_t)mat * CDIM * CDIM;
  const float* bias = (mat == 0) ? bq : (mat == 1) ? bk : bv;
  if (threadIdx.x < 32) tdm_load_64k(Wm + (size_t)cbase * CDIM, sW);
  __syncthreads();

  v8f acc[4] = {};
  const __bf16* arow = A + (size_t)ar * CDIM;
  v16bf a_cur = frag_cat(arow + 8 * half, arow + 8 * half + 16);
  for (int t = 0; t < 16; ++t) {
    v16bf a_nxt = a_cur;
    if (t < 15) {
      const int ka = 32 * (t + 1) + 8 * half;
      a_nxt = frag_cat(arow + ka, arow + ka + 16);
    }
    const int kb = 32 * t + 8 * half;
    #pragma unroll
    for (int nf = 0; nf < 4; ++nf) {
      acc[nf] = wmma_bf16(a_cur, frag16(&sW[(size_t)(nf * 16 + ar) * CDIM + kb]),
                          acc[nf]);
    }
    a_cur = a_nxt;
  }
  #pragma unroll
  for (int nf = 0; nf < 4; ++nf) {
    const int c = cbase + nf * 16 + ar;
    const float bb = bias[c];
    #pragma unroll
    for (int f = 0; f < 8; ++f) {
      const int row = r0 + 8 * half + f;
      const __bf16 o = (__bf16)(acc[nf][f] + bb);
      if (mat == 2)       Vtg[((size_t)b * CDIM + c) * NDIM + row] = o;
      else if (mat == 0)  Qg[((size_t)b * NDIM + row) * CDIM + c] = o;
      else                Kg[((size_t)b * NDIM + row) * CDIM + c] = o;
    }
  }
}

// ---------------------------------------------------------------------------
// Kernel 4: flash attention. Block = 16 query rows, 8 waves.
// Per 128-key block: wave w scores keys [j+16w,+16) (16 WMMAs), online
// softmax with LDS cross-wave max/sum exchange, P staged in LDS as bf16,
// then wave w accumulates channels [64w,+64) of O (16 WMMAs vs Vt).
// K/V fragments prefetched one step ahead.
// ---------------------------------------------------------------------------
__global__ __launch_bounds__(256) void flash_attn(
    const __bf16* __restrict__ Qg, const __bf16* __restrict__ Kg,
    const __bf16* __restrict__ Vt, __bf16* __restrict__ Og) {
  __shared__ __bf16 sQ[16 * CDIM];     // 16 KB
  __shared__ __bf16 sP[16 * 128];      //  4 KB
  __shared__ float pmax[8][16];
  __shared__ float psum[8][16];
  const int b  = blockIdx.y;
  const int r0 = blockIdx.x * 16;
  const int tid = threadIdx.x, lane = tid & 31, w = tid >> 5;
  const int ar = lane & 15;
  const int half = lane >> 4;
  const int rhalf = 8 * half;
  const int cb = w * 64;
  const float scale = 0.044194173824159216f;   // 1/sqrt(512)
  { // stage Q tile into LDS (1024 x 16B)
    const uint4* src = (const uint4*)(Qg + ((size_t)b * NDIM + r0) * CDIM);
    uint4* dst = (uint4*)sQ;
    for (int i = tid; i < 16 * CDIM * 2 / 16; i += 256) dst[i] = src[i];
  }
  __syncthreads();
  float mloc[8], lloc[8];
  #pragma unroll
  for (int f = 0; f < 8; ++f) { mloc[f] = -1e30f; lloc[f] = 0.f; }
  v8f O[4] = {};

  for (int j = 0; j < NDIM; j += 128) {
    // --- scores: S = scale * Q (16xC) . K_slice^T (Cx16) ---
    v8f S = {};
    const __bf16* Krow = Kg + ((size_t)b * NDIM + (j + 16 * w + ar)) * CDIM;
    v16bf kf = frag16(Krow + 8 * half);
    #pragma unroll
    for (int t = 0; t < 16; ++t) {
      v16bf kf_nxt = kf;
      if (t < 15) kf_nxt = frag16(Krow + 32 * (t + 1) + 8 * half);
      const int ka = 32 * t + 8 * half;
      v16bf a = frag_cat(&sQ[ar * CDIM + ka], &sQ[ar * CDIM + ka + 16]);
      S = wmma_bf16(a, kf, S);
      kf = kf_nxt;
    }
    // --- row max within this wave's 16-key tile ---
    float rm[8];
    #pragma unroll
    for (int f = 0; f < 8; ++f) {
      float v = S[f] * scale;
      S[f] = v;
      #pragma unroll
      for (int m = 1; m < 16; m <<= 1) v = fmaxf(v, __shfl_xor(v, m, 32));
      rm[f] = v;
    }
    if (ar == 0) {
      #pragma unroll
      for (int f = 0; f < 8; ++f) pmax[w][rhalf + f] = rm[f];
    }
    __syncthreads();
    // --- combine maxima, exponentiate, stage P, partial row sums ---
    float alpha[8], rs[8];
    #pragma unroll
    for (int f = 0; f < 8; ++f) {
      float bm = -1e30f;
      #pragma unroll
      for (int ww = 0; ww < 8; ++ww) bm = fmaxf(bm, pmax[ww][rhalf + f]);
      const float mnew = fmaxf(mloc[f], bm);
      alpha[f] = __expf(mloc[f] - mnew);
      mloc[f] = mnew;
      const float p = __expf(S[f] - mnew);
      sP[(rhalf + f) * 128 + w * 16 + ar] = (__bf16)p;
      float sum = p;
      #pragma unroll
      for (int m = 1; m < 16; m <<= 1) sum += __shfl_xor(sum, m, 32);
      rs[f] = sum;
    }
    if (ar == 0) {
      #pragma unroll
      for (int f = 0; f < 8; ++f) psum[w][rhalf + f] = rs[f];
    }
    __syncthreads();
    // --- update l, rescale O ---
    #pragma unroll
    for (int f = 0; f < 8; ++f) {
      float bs = 0.f;
      #pragma unroll
      for (int ww = 0; ww < 8; ++ww) bs += psum[ww][rhalf + f];
      lloc[f] = lloc[f] * alpha[f] + bs;
      const float a_ = alpha[f];
      O[0][f] *= a_; O[1][f] *= a_; O[2][f] *= a_; O[3][f] *= a_;
    }
    // --- O += P (16x128) . V[j:j+128, cb:cb+64]  (Vt is [c][n]) ---
    const __bf16* Vbase = Vt + (size_t)b * CDIM * NDIM + (size_t)(cb + ar) * NDIM
                          + j + 8 * half;
    v16bf vf = frag16(Vbase);
    #pragma unroll
    for (int kt = 0; kt < 4; ++kt) {
      const int ka = 32 * kt + 8 * half;
      v16bf p = frag_cat(&sP[ar * 128 + ka], &sP[ar * 128 + ka + 16]);
      #pragma unroll
      for (int nf = 0; nf < 4; ++nf) {
        v16bf vf_nxt = vf;
        if (kt * 4 + nf < 15) {
          const int nf2 = (nf + 1) & 3;
          const int kt2 = kt + ((nf + 1) >> 2);
          vf_nxt = frag16(Vbase + (size_t)(nf2 * 16) * NDIM + 32 * kt2);
        }
        O[nf] = wmma_bf16(p, vf, O[nf]);
        vf = vf_nxt;
      }
    }
  }
  // --- epilogue: O /= l, store bf16 [b][n][c] ---
  #pragma unroll
  for (int f = 0; f < 8; ++f) {
    const int row = r0 + rhalf + f;
    const float inv = 1.0f / lloc[f];
    __bf16* orow = Og + ((size_t)b * NDIM + row) * CDIM + cb + ar;
    orow[0]  = (__bf16)(O[0][f] * inv);
    orow[16] = (__bf16)(O[1][f] * inv);
    orow[32] = (__bf16)(O[2][f] * inv);
    orow[48] = (__bf16)(O[3][f] * inv);
  }
}

// ---------------------------------------------------------------------------
// Kernel 5: out = O @ Wo + bo + residual (fp32). Same TDM-staged weight tile.
// ---------------------------------------------------------------------------
__global__ __launch_bounds__(256) void out_proj(
    const __bf16* __restrict__ Og, const __bf16* __restrict__ wt_o,
    const float* __restrict__ bo, const float* __restrict__ xin,
    float* __restrict__ out) {
  __shared__ __bf16 sW[64 * CDIM];     // 64 KB weight tile
  const int b = blockIdx.z;
  const int lane = threadIdx.x & 31, w = threadIdx.x >> 5;
  const int r0 = blockIdx.x * 128 + w * 16;
  const int cbase = blockIdx.y * 64;
  const int ar = lane & 15;
  const int half = lane >> 4;
  const __bf16* A = Og + ((size_t)b * NDIM + r0) * CDIM;
  if (threadIdx.x < 32) tdm_load_64k(wt_o + (size_t)cbase * CDIM, sW);
  __syncthreads();

  v8f acc[4] = {};
  const __bf16* arow = A + (size_t)ar * CDIM;
  v16bf a_cur = frag_cat(arow + 8 * half, arow + 8 * half + 16);
  for (int t = 0; t < 16; ++t) {
    v16bf a_nxt = a_cur;
    if (t < 15) {
      const int ka = 32 * (t + 1) + 8 * half;
      a_nxt = frag_cat(arow + ka, arow + ka + 16);
    }
    const int kb = 32 * t + 8 * half;
    #pragma unroll
    for (int nf = 0; nf < 4; ++nf) {
      acc[nf] = wmma_bf16(a_cur, frag16(&sW[(size_t)(nf * 16 + ar) * CDIM + kb]),
                          acc[nf]);
    }
    a_cur = a_nxt;
  }
  #pragma unroll
  for (int nf = 0; nf < 4; ++nf) {
    const int c = cbase + nf * 16 + ar;
    const float bb = bo[c];
    #pragma unroll
    for (int f = 0; f < 8; ++f) {
      const size_t idx = ((size_t)b * NDIM + r0 + 8 * half + f) * CDIM + c;
      out[idx] = acc[nf][f] + bb + xin[idx];
    }
  }
}

// ---------------------------------------------------------------------------
// Launch
// ---------------------------------------------------------------------------
extern "C" void kernel_launch(void* const* d_in, const int* in_sizes, int n_in,
                              void* d_out, int out_size, void* d_ws, size_t ws_size,
                              hipStream_t stream) {
  const float* x     = (const float*)d_in[0];
  const float* gamma = (const float*)d_in[1];
  const float* beta  = (const float*)d_in[2];
  const float* Wq = (const float*)d_in[3];  const float* bq = (const float*)d_in[4];
  const float* Wk = (const float*)d_in[5];  const float* bk = (const float*)d_in[6];
  const float* Wv = (const float*)d_in[7];  const float* bv = (const float*)d_in[8];
  const float* Wo = (const float*)d_in[9];  const float* bo = (const float*)d_in[10];

  char* ws = (char*)d_ws;
  const size_t SZ_ACT = (size_t)BATCH * NDIM * CDIM * sizeof(__bf16);   // 16 MiB
  const size_t SZ_WT  = (size_t)4 * CDIM * CDIM * sizeof(__bf16);       //  2 MiB
  __bf16* h   = (__bf16*)(ws);
  __bf16* wt  = (__bf16*)(ws + SZ_ACT);
  __bf16* Qg  = (__bf16*)(ws + SZ_ACT + SZ_WT);
  __bf16* Kg  = (__bf16*)(ws + 2 * SZ_ACT + SZ_WT);
  __bf16* Vt  = (__bf16*)(ws + 3 * SZ_ACT + SZ_WT);
  __bf16* Ob  = (__bf16*)(ws + 4 * SZ_ACT + SZ_WT);
  float*  out = (float*)d_out;

  prep_weights<<<dim3(CDIM * CDIM / 256, 4), 256, 0, stream>>>(Wq, Wk, Wv, Wo, wt);
  group_norm_k<<<dim3(BATCH * 32), 256, 0, stream>>>(x, gamma, beta, h);
  qkv_gemm<<<dim3(NDIM / 128, CDIM / 64, 3 * BATCH), 256, 0, stream>>>(
      h, wt, bq, bk, bv, Qg, Kg, Vt);
  flash_attn<<<dim3(NDIM / 16, BATCH), 256, 0, stream>>>(Qg, Kg, Vt, Ob);
  out_proj<<<dim3(NDIM / 128, CDIM / 64, BATCH), 256, 0, stream>>>(
      Ob, wt + (size_t)3 * CDIM * CDIM, bo, x, out);
}
